// SkipGramModel_21225728377105
// MI455X (gfx1250) — matrix operand order repaired and checked by reading, not compiled, not run on hardware
//
#include <hip/hip_runtime.h>
#include <hip/hip_bf16.h>

typedef __attribute__((ext_vector_type(16))) _Float16 v16h;
typedef __attribute__((ext_vector_type(8)))  _Float16 v8h;
typedef __attribute__((ext_vector_type(8)))  float    v8f;

#define NROWS 2560          // BATCH * WALK_LEN
#define DIM   128
#define NPOS  24640
#define NNEG  123200
#define POS_GROUPS (NPOS / 16)            // 1540
#define NEG_GROUPS (NNEG / 16)            // 7700
#define TOT_GROUPS (POS_GROUPS + NEG_GROUPS)  // 9240
#define WAVES_PER_BLOCK 8
#define NBLOCKS (TOT_GROUPS / WAVES_PER_BLOCK) // 1155 (exact)
#define CLIPV 6.0f

// ---------------------------------------------------------------------------
// Kernel 1: gather u_emb[nodes], v_emb[nodes] -> compact f16 buffers in d_ws.
// 2560 rows x 128 cols x 2 tables; coalesced (128 consecutive lanes per row).
// ---------------------------------------------------------------------------
__global__ void __launch_bounds__(256)
gather_f16(const float* __restrict__ u, const float* __restrict__ v,
           const int* __restrict__ nodes,
           _Float16* __restrict__ gu, _Float16* __restrict__ gv) {
  int tid = blockIdx.x * blockDim.x + threadIdx.x;
  if (tid >= NROWS * DIM) return;
  int row = tid >> 7;
  int col = tid & (DIM - 1);
  size_t src = (size_t)nodes[row] * DIM + col;
  gu[tid] = (_Float16)u[src];
  gv[tid] = (_Float16)v[src];
}

// ---------------------------------------------------------------------------
// Kernel 2: one wave per group of 16 pairs. Diagonal of a 16x16 WMMA tile
// gives the 16 dot products (A rows = gathered u vectors, B cols = gathered
// v vectors). K=128 done as 4 chained v_wmma_f32_16x16x32_f16.
// ---------------------------------------------------------------------------
__global__ void __launch_bounds__(256)
score_kernel(const _Float16* __restrict__ gu, const _Float16* __restrict__ gv,
             const int* __restrict__ posu, const int* __restrict__ posv,
             const int* __restrict__ negu, const int* __restrict__ negv,
             float* __restrict__ partials) {
  __shared__ float sred[WAVES_PER_BLOCK];

  const int wave = threadIdx.x >> 5;
  const int lane = threadIdx.x & 31;
  const int g    = blockIdx.x * WAVES_PER_BLOCK + wave;   // [0, TOT_GROUPS)
  const int r    = lane & 15;   // row (A) / col (B) index within the tile
  const int half = lane >> 4;   // lane half selects K sub-block

  const bool isNeg = (g >= POS_GROUPS);
  const int* __restrict__ iu = isNeg ? negu : posu;
  const int* __restrict__ iv = isNeg ? negv : posv;
  const int  base = (isNeg ? (g - POS_GROUPS) : g) * 16;

  // Per-lane row/col gather indices (values in [0, NROWS))
  const int ru = iu[base + r];
  const int rv = iv[base + r];
  const _Float16* __restrict__ arow = gu + (size_t)ru * DIM;
  const _Float16* __restrict__ bcol = gv + (size_t)rv * DIM;

  v8f c = {};
#pragma unroll
  for (int kc = 0; kc < 4; ++kc) {
    const int kb = kc * 32;
    // A 16x32 f16 layout: lane(l<16): row l, K = kb+0..15 (contiguous)
    //                     lane(l>=16): row l-16, K = kb+16..31
    v16h a = *(const v16h*)(arow + kb + half * 16);       // 32B aligned
    // B 32x16 f16 layout: lane(l<16): col l, K = kb+0..15 (contiguous)
    //                     lane(l>=16): col l-16, K = kb+16..31
    v16h b = *(const v16h*)(bcol + kb + half * 16);       // 32B aligned
    c = __builtin_amdgcn_wmma_f32_16x16x32_f16(
        /*neg_a=*/false, a, /*neg_b=*/false, b,
        /*c_mod=*/(short)0, c, /*reuse_a=*/false, /*reuse_b=*/false);
  }

  // Diagonal D[j][j]: VGPR j @ lane j (pairs 0..7), VGPR j @ lane j+24 (8..15)
  float s = 0.0f;
#pragma unroll
  for (int j = 0; j < 8; ++j)
    s = (lane == j || lane == j + 24) ? c[j] : s;

  float contrib = 0.0f;
  if (lane < 8 || lane >= 24) {
    s = fminf(fmaxf(s, -CLIPV), CLIPV);
    // loss contribution: pos -> log(1+exp(-s)); neg -> log(1+exp(+s))
    float x = isNeg ? s : -s;
    contrib = __logf(1.0f + __expf(x));
  }

  // wave32 butterfly reduction (deterministic)
#pragma unroll
  for (int m = 16; m; m >>= 1) contrib += __shfl_xor(contrib, m, 32);

  if (lane == 0) sred[wave] = contrib;
  __syncthreads();
  if (threadIdx.x == 0) {
    float t = 0.0f;
#pragma unroll
    for (int i = 0; i < WAVES_PER_BLOCK; ++i) t += sred[i];
    partials[blockIdx.x] = t;
  }
}

// ---------------------------------------------------------------------------
// Kernel 3: deterministic final reduction of 1155 per-block partials.
// ---------------------------------------------------------------------------
__global__ void __launch_bounds__(256)
reduce_kernel(const float* __restrict__ partials, float* __restrict__ out) {
  __shared__ float s[256];
  float t = 0.0f;
  for (int i = threadIdx.x; i < NBLOCKS; i += 256) t += partials[i];
  s[threadIdx.x] = t;
  __syncthreads();
#pragma unroll
  for (int m = 128; m; m >>= 1) {
    if (threadIdx.x < m) s[threadIdx.x] += s[threadIdx.x + m];
    __syncthreads();
  }
  if (threadIdx.x == 0) out[0] = s[0];
}

// ---------------------------------------------------------------------------
extern "C" void kernel_launch(void* const* d_in, const int* in_sizes, int n_in,
                              void* d_out, int out_size, void* d_ws, size_t ws_size,
                              hipStream_t stream) {
  const float* u_emb = (const float*)d_in[0];
  const float* v_emb = (const float*)d_in[1];
  const int*   nodes = (const int*)d_in[2];
  const int*   posu  = (const int*)d_in[3];
  const int*   posv  = (const int*)d_in[4];
  const int*   negu  = (const int*)d_in[5];
  const int*   negv  = (const int*)d_in[6];

  char* ws = (char*)d_ws;
  const size_t gbytes = (size_t)NROWS * DIM * sizeof(_Float16);  // 655,360 B
  _Float16* gu = (_Float16*)ws;
  _Float16* gv = (_Float16*)(ws + gbytes);
  float* partials = (float*)(ws + 2 * gbytes);   // + 4620 B

  gather_f16<<<(NROWS * DIM) / 256, 256, 0, stream>>>(u_emb, v_emb, nodes, gu, gv);
  score_kernel<<<NBLOCKS, 256, 0, stream>>>(gu, gv, posu, posv, negu, negv, partials);
  reduce_kernel<<<1, 256, 0, stream>>>(partials, (float*)d_out);
}
